// DisentangledLAISelfAttention_2465311228235
// MI455X (gfx1250) — compile-verified, matrix operand order checked by reading, not run on hardware
//
#include <hip/hip_runtime.h>
#include <hip/hip_bf16.h>

// ---------------------------------------------------------------------------
// CDNA5 (gfx1250) disentangled attention, all matmuls via v_wmma_f32_16x16x32_bf16
// ---------------------------------------------------------------------------

typedef __attribute__((ext_vector_type(16))) __bf16 v16bf;
typedef __attribute__((ext_vector_type(8)))  float  v8f;

#define WMMA_BF16(a, b, c) \
  __builtin_amdgcn_wmma_f32_16x16x32_bf16(false, (a), false, (b), (short)0, (c), false, false)

__device__ __forceinline__ unsigned short f2bf(float f) {
  unsigned int u = __float_as_uint(f);
  u += 0x7FFFu + ((u >> 16) & 1u);          // round-to-nearest-even
  return (unsigned short)(u >> 16);
}

union Frag16 {
  v16bf v;
  uint4 q[2];
  unsigned short u[16];
};

// 16 bf16 elements per lane: elems 0..7 from p0 (contiguous), 8..15 from p1.
__device__ __forceinline__ v16bf load_frag_bf16(const unsigned short* p0,
                                                const unsigned short* p1) {
  Frag16 f;
  f.q[0] = *(const uint4*)p0;
  f.q[1] = *(const uint4*)p1;
  return f.v;
}

// ---------------------------------------------------------------------------
// Kernel 1a: transpose + bf16-convert a 512x512 weight: Wt[o][i] = bf16(W[i][o])
// ---------------------------------------------------------------------------
__global__ void __launch_bounds__(256)
wconv_kernel(const float* __restrict__ W, unsigned short* __restrict__ Wt) {
  int i = blockIdx.x * blockDim.x + threadIdx.x;   // 512*512 total
  int o  = i >> 9;
  int ci = i & 511;
  Wt[i] = f2bf(W[(ci << 9) | o]);
}

// ---------------------------------------------------------------------------
// Kernel 1b: elementwise bf16 conversion of activations (memory-bound; keeps
// all f32->bf16 VALU work OUT of the GEMM inner loops).
// ---------------------------------------------------------------------------
__global__ void __launch_bounds__(256)
xconv_kernel(const float* __restrict__ X, unsigned short* __restrict__ Xb, int n4) {
  int i = blockIdx.x * blockDim.x + threadIdx.x;   // n/4 total
  if (i >= n4) return;
  float4 x = ((const float4*)X)[i];
  ushort4 y;
  y.x = f2bf(x.x); y.y = f2bf(x.y); y.z = f2bf(x.z); y.w = f2bf(x.w);
  ((ushort4*)Xb)[i] = y;
}

// ---------------------------------------------------------------------------
// Kernel 2: projection GEMM  Out = bf16(Xb[M,512] @ Wt^T + bias)
// Xb, Wt are bf16. One wave -> 16 rows x 128 cols (8 WMMA tiles), pure
// load_b128 + wmma inner loop (A fragment reused across 8 column tiles).
// vmode==1 stores V transposed: Out[b][col][n] with b = row>>10, n = row&1023.
// ---------------------------------------------------------------------------
__global__ void __launch_bounds__(128)
proj_gemm_kernel(const unsigned short* __restrict__ Xb,
                 const unsigned short* __restrict__ Wt,
                 const float* __restrict__ bias,
                 unsigned short* __restrict__ Out,
                 int M, int vmode) {
  const int wave = blockIdx.x * (blockDim.x >> 5) + (threadIdx.x >> 5);
  const int lane = threadIdx.x & 31;
  const int lr   = lane & 15;
  const int half = lane >> 4;
  const int rowTile = wave >> 2;       // 4 column groups of 128
  const int colGrp  = wave & 3;
  if (rowTile * 16 >= M) return;
  const int row = rowTile * 16 + lr;
  const int n0  = colGrp * 128;

  v8f acc[8];
  #pragma unroll
  for (int t = 0; t < 8; ++t)
    #pragma unroll
    for (int e = 0; e < 8; ++e) acc[t][e] = 0.0f;

  const unsigned short* xrow = Xb + row * 512;
  for (int kk = 0; kk < 512; kk += 32) {
    v16bf aF = load_frag_bf16(xrow + kk + half * 8, xrow + kk + 16 + half * 8);
    #pragma unroll
    for (int t = 0; t < 8; ++t) {
      const int col = n0 + t * 16 + lr;
      const unsigned short* wrow = Wt + col * 512 + kk;
      v16bf bF = load_frag_bf16(wrow + half * 8, wrow + 16 + half * 8);
      acc[t] = WMMA_BF16(aF, bF, acc[t]);
    }
  }

  #pragma unroll
  for (int t = 0; t < 8; ++t) {
    const int col = n0 + t * 16 + lr;
    const float bv = bias[col];
    #pragma unroll
    for (int v = 0; v < 8; ++v) {
      const int r = rowTile * 16 + v + 8 * half;       // C-layout row
      const unsigned short h16 = f2bf(acc[t][v] + bv);
      if (vmode) {
        Out[(r >> 10) * 524288 + col * 1024 + (r & 1023)] = h16;
      } else {
        Out[r * 512 + col] = h16;
      }
    }
  }
}

// ---------------------------------------------------------------------------
// Kernel 3: fused disentangled attention.
// One wave handles one (b, h, 16-query tile); k processed in blocks of 32.
//   S[q,k] = (Q·K + Q·posK[clip(q-k+512)] + K·posQ[clip(q-k+512)]) / sqrt(192)
// c2p/p2c computed as 48-wide position-window WMMAs, gathered through LDS.
// Online softmax, P (bf16 through LDS) x V WMMA with K=32.
// ---------------------------------------------------------------------------
__global__ void __launch_bounds__(128)
attn_kernel(const unsigned short* __restrict__ Qb,   // [B*N][512] bf16
            const unsigned short* __restrict__ Kb,   // [B*N][512] bf16
            const unsigned short* __restrict__ Vt,   // [B][512][1024] bf16 (transposed)
            const unsigned short* __restrict__ posK, // [1024][512] bf16
            const unsigned short* __restrict__ posQ, // [1024][512] bf16
            float* __restrict__ out) {               // [B*N][512] f32
  __shared__ __align__(16) float  w48[4][16][48];
  __shared__ __align__(16) float  v48s[4][32][48];
  __shared__ __align__(16) unsigned short plds[4][16][32];

  const int wv   = threadIdx.x >> 5;
  const int lane = threadIdx.x & 31;
  const int lr   = lane & 15;
  const int half = lane >> 4;
  const int wave = blockIdx.x * 4 + wv;     // 4096 waves total
  const int qt = wave & 63;
  const int bh = wave >> 6;
  const int b  = bh >> 3;
  const int h  = bh & 7;
  const int q0 = qt << 4;
  const float rscale = 0.07216878364870322f;   // 1/sqrt(192)

  // persistent Q A-fragments (d = 64 -> 2 chunks of 32)
  const unsigned short* qrow = Qb + (b * 1024 + q0 + lr) * 512 + h * 64;
  v16bf aQ[2];
  aQ[0] = load_frag_bf16(qrow + half * 8,      qrow + 16 + half * 8);
  aQ[1] = load_frag_bf16(qrow + 32 + half * 8, qrow + 48 + half * 8);

  float m[8], l[8];
  v8f O[4];
  #pragma unroll
  for (int v = 0; v < 8; ++v) { m[v] = -1e30f; l[v] = 0.0f; }
  #pragma unroll
  for (int t = 0; t < 4; ++t)
    #pragma unroll
    for (int e = 0; e < 8; ++e) O[t][e] = 0.0f;

  #pragma unroll 1
  for (int kb = 0; kb < 32; ++kb) {
    const int k0 = kb * 32;

    const unsigned short* krow0 = Kb + (b * 1024 + k0 + lr) * 512 + h * 64;
    const unsigned short* krow1 = krow0 + 16 * 512;

    // prefetch next k-block's K rows and V tile (global_prefetch_b8)
    if (kb < 31) {
      __builtin_prefetch(krow0 + 32 * 512, 0, 3);
      __builtin_prefetch(krow1 + 32 * 512, 0, 3);
      __builtin_prefetch(Vt + b * 524288 + (h * 64 + lr) * 1024 + k0 + 32, 0, 3);
    }

    // ---- c2c: S[16q x 32k] as two 16x16 tiles ----
    v8f S0, S1;
    #pragma unroll
    for (int e = 0; e < 8; ++e) { S0[e] = 0.0f; S1[e] = 0.0f; }
    #pragma unroll
    for (int c = 0; c < 2; ++c) {
      v16bf bK0 = load_frag_bf16(krow0 + c * 32 + half * 8, krow0 + c * 32 + 16 + half * 8);
      v16bf bK1 = load_frag_bf16(krow1 + c * 32 + half * 8, krow1 + c * 32 + 16 + half * 8);
      S0 = WMMA_BF16(aQ[c], bK0, S0);
      S1 = WMMA_BF16(aQ[c], bK1, S1);
    }

    // window of relative positions covering this (q-tile, k-block)
    const int basep = q0 - k0 + 481;   // delta = basep + t, t in [0,46]

    // ---- c2p: W48[16q x 48] = Q x posK_window ----
    #pragma unroll
    for (int j0 = 0; j0 < 48; j0 += 16) {
      int p = basep + j0 + lr;
      p = p < 0 ? 0 : (p > 1023 ? 1023 : p);
      const unsigned short* prow = posK + p * 512 + h * 64;
      v8f acc;
      #pragma unroll
      for (int e = 0; e < 8; ++e) acc[e] = 0.0f;
      #pragma unroll
      for (int c = 0; c < 2; ++c) {
        v16bf bP = load_frag_bf16(prow + c * 32 + half * 8, prow + c * 32 + 16 + half * 8);
        acc = WMMA_BF16(aQ[c], bP, acc);
      }
      #pragma unroll
      for (int v = 0; v < 8; ++v) w48[wv][v + 8 * half][j0 + lr] = acc[v];
    }

    // ---- p2c: V48[32k x 48] = K_tile x posQ_window ----
    v16bf aK[2][2];
    #pragma unroll
    for (int rt = 0; rt < 2; ++rt) {
      const unsigned short* kr = Kb + (b * 1024 + k0 + rt * 16 + lr) * 512 + h * 64;
      aK[rt][0] = load_frag_bf16(kr + half * 8,      kr + 16 + half * 8);
      aK[rt][1] = load_frag_bf16(kr + 32 + half * 8, kr + 48 + half * 8);
    }
    #pragma unroll
    for (int j0 = 0; j0 < 48; j0 += 16) {
      int p = basep + j0 + lr;
      p = p < 0 ? 0 : (p > 1023 ? 1023 : p);
      const unsigned short* prow = posQ + p * 512 + h * 64;
      v16bf bP0 = load_frag_bf16(prow + half * 8,      prow + 16 + half * 8);
      v16bf bP1 = load_frag_bf16(prow + 32 + half * 8, prow + 48 + half * 8);
      #pragma unroll
      for (int rt = 0; rt < 2; ++rt) {
        v8f acc;
        #pragma unroll
        for (int e = 0; e < 8; ++e) acc[e] = 0.0f;
        acc = WMMA_BF16(aK[rt][0], bP0, acc);
        acc = WMMA_BF16(aK[rt][1], bP1, acc);
        #pragma unroll
        for (int v = 0; v < 8; ++v) v48s[wv][rt * 16 + v + 8 * half][j0 + lr] = acc[v];
      }
    }
    __builtin_amdgcn_wave_barrier();   // LDS in-order within wave; block motion

    // ---- gather Toeplitz windows + scale ----
    #pragma unroll
    for (int v = 0; v < 8; ++v) {
      const int qr = v + 8 * half;
      const int t0 = qr - lr + 31;   // keys k0 + lr      (krel = lr)
      const int t1 = qr - lr + 15;   // keys k0 + 16 + lr (krel = 16+lr)
      S0[v] = (S0[v] + w48[wv][qr][t0] + v48s[wv][lr][t0]) * rscale;
      S1[v] = (S1[v] + w48[wv][qr][t1] + v48s[wv][16 + lr][t1]) * rscale;
    }

    // ---- online softmax over 32 keys (row = 16-lane half reduction) ----
    #pragma unroll
    for (int v = 0; v < 8; ++v) {
      float mx = fmaxf(S0[v], S1[v]);
      mx = fmaxf(mx, __shfl_xor(mx, 1, 32));
      mx = fmaxf(mx, __shfl_xor(mx, 2, 32));
      mx = fmaxf(mx, __shfl_xor(mx, 4, 32));
      mx = fmaxf(mx, __shfl_xor(mx, 8, 32));
      const float mn    = fmaxf(m[v], mx);
      const float alpha = __expf(m[v] - mn);
      const float p0 = __expf(S0[v] - mn);
      const float p1 = __expf(S1[v] - mn);
      float rs = p0 + p1;
      rs += __shfl_xor(rs, 1, 32);
      rs += __shfl_xor(rs, 2, 32);
      rs += __shfl_xor(rs, 4, 32);
      rs += __shfl_xor(rs, 8, 32);
      l[v] = l[v] * alpha + rs;
      m[v] = mn;
      #pragma unroll
      for (int t = 0; t < 4; ++t) O[t][v] *= alpha;
      const int qr = v + 8 * half;
      plds[wv][qr][lr]      = f2bf(p0);
      plds[wv][qr][16 + lr] = f2bf(p1);
    }
    __builtin_amdgcn_wave_barrier();

    // ---- O += P[16x32] x V[32x64] (4 WMMA, K=32) ----
    {
      const unsigned short* pr = &plds[wv][lr][0];
      v16bf aP = load_frag_bf16(pr + half * 8, pr + 16 + half * 8);
      const unsigned short* vbase = Vt + b * 524288 + (h * 64) * 1024 + k0;
      #pragma unroll
      for (int t = 0; t < 4; ++t) {
        const unsigned short* vr = vbase + (t * 16 + lr) * 1024;
        v16bf bV = load_frag_bf16(vr + half * 8, vr + 16 + half * 8);
        O[t] = WMMA_BF16(aP, bV, O[t]);
      }
    }
  }

  // ---- epilogue: normalize and store fp32 [B,N,C] ----
  float* orow = out + (b * 1024 + q0) * 512 + h * 64;
  #pragma unroll
  for (int v = 0; v < 8; ++v) {
    const int qr = v + 8 * half;
    const float inv = 1.0f / l[v];
    #pragma unroll
    for (int t = 0; t < 4; ++t) {
      orow[qr * 512 + t * 16 + lr] = O[t][v] * inv;
    }
  }
}

// ---------------------------------------------------------------------------
// Launch
// ---------------------------------------------------------------------------
extern "C" void kernel_launch(void* const* d_in, const int* in_sizes, int n_in,
                              void* d_out, int out_size, void* d_ws, size_t ws_size,
                              hipStream_t stream) {
  const float* q   = (const float*)d_in[0];
  const float* k   = (const float*)d_in[1];
  const float* val = (const float*)d_in[2];
  const float* rel = (const float*)d_in[3];
  const float* Wq  = (const float*)d_in[4];
  const float* bq  = (const float*)d_in[5];
  const float* Wk  = (const float*)d_in[6];
  const float* bk  = (const float*)d_in[7];
  const float* Wv  = (const float*)d_in[8];
  const float* bv  = (const float*)d_in[9];
  float* out = (float*)d_out;

  char* ws = (char*)d_ws;
  unsigned short* Wqt  = (unsigned short*)ws; ws += 512 * 512 * 2;
  unsigned short* Wkt  = (unsigned short*)ws; ws += 512 * 512 * 2;
  unsigned short* Wvt  = (unsigned short*)ws; ws += 512 * 512 * 2;
  unsigned short* Qb   = (unsigned short*)ws; ws += 8192 * 512 * 2;
  unsigned short* Kb   = (unsigned short*)ws; ws += 8192 * 512 * 2;
  unsigned short* Vt   = (unsigned short*)ws; ws += 8192 * 512 * 2;
  unsigned short* pK   = (unsigned short*)ws; ws += 1024 * 512 * 2;
  unsigned short* pQ   = (unsigned short*)ws; ws += 1024 * 512 * 2;
  unsigned short* qbf  = (unsigned short*)ws; ws += 8192 * 512 * 2;
  unsigned short* kbf  = (unsigned short*)ws; ws += 8192 * 512 * 2;
  unsigned short* vbf  = (unsigned short*)ws; ws += 8192 * 512 * 2;
  unsigned short* rbf  = (unsigned short*)ws; ws += 1024 * 512 * 2;

  // bf16 weight transposes
  wconv_kernel<<<1024, 256, 0, stream>>>(Wq, Wqt);
  wconv_kernel<<<1024, 256, 0, stream>>>(Wk, Wkt);
  wconv_kernel<<<1024, 256, 0, stream>>>(Wv, Wvt);

  // bf16 activation copies (float4-vectorized)
  const int nBig4 = 8192 * 512 / 4;   // 1048576
  const int nRel4 = 1024 * 512 / 4;   // 131072
  xconv_kernel<<<nBig4 / 256, 256, 0, stream>>>(q,   qbf, nBig4);
  xconv_kernel<<<nBig4 / 256, 256, 0, stream>>>(k,   kbf, nBig4);
  xconv_kernel<<<nBig4 / 256, 256, 0, stream>>>(val, vbf, nBig4);
  xconv_kernel<<<nRel4 / 256, 256, 0, stream>>>(rel, rbf, nRel4);

  // projections (all-WMMA)
  proj_gemm_kernel<<<512, 128, 0, stream>>>(qbf, Wqt, bq, Qb, 8192, 0);
  proj_gemm_kernel<<<512, 128, 0, stream>>>(kbf, Wkt, bk, Kb, 8192, 0);
  proj_gemm_kernel<<<512, 128, 0, stream>>>(vbf, Wvt, bv, Vt, 8192, 1);
  proj_gemm_kernel<<<64,  128, 0, stream>>>(rbf, Wkt, bk, pK, 1024, 0);
  proj_gemm_kernel<<<64,  128, 0, stream>>>(rbf, Wqt, bq, pQ, 1024, 0);

  // fused attention (WMMA c2c/c2p/p2c/PV + online softmax)
  attn_kernel<<<1024, 128, 0, stream>>>(Qb, Kb, Vt, pK, pQ, out);
}
